// Attention_24988119728323
// MI455X (gfx1250) — matrix operand order repaired and checked by reading, not compiled, not run on hardware
//
#include <hip/hip_runtime.h>

typedef __attribute__((ext_vector_type(16))) _Float16 v16h;
typedef __attribute__((ext_vector_type(8)))  _Float16 v8h;
typedef __attribute__((ext_vector_type(4)))  _Float16 v4h;
typedef __attribute__((ext_vector_type(8)))  float    v8f;

namespace {
constexpr int kNB = 2, kH = 16, kS = 2048, kDH = 64;
constexpr int BM = 64;   // query rows per workgroup (4 waves x 16)
constexpr int BN = 64;   // keys per main-loop iteration
constexpr int KP = 72;   // K-tile LDS row stride (halves), padded
constexpr int VP = 72;   // V^T-tile LDS row stride (halves), padded
constexpr int PP = 72;   // P^T-tile LDS row stride (halves), padded
// softmax runs in log2 domain: fold (1/sqrt(64)) * log2(e) into Q
constexpr float QSCALE = 0.125f * 1.44269504088896340736f;
constexpr float MASK_FILL_LOG2 = -10000.0f * 1.44269504088896340736f;
}

__device__ __forceinline__ v16h cat8(v8h lo, v8h hi) {
  return __builtin_shufflevector(lo, hi, 0, 1, 2, 3, 4, 5, 6, 7,
                                 8, 9, 10, 11, 12, 13, 14, 15);
}

__device__ __forceinline__ v8f wmma_f16(v16h a, v16h b, v8f c) {
  return __builtin_amdgcn_wmma_f32_16x16x32_f16(false, a, false, b,
                                                (short)0, c, false, false);
}

__global__ __launch_bounds__(128)
void flash_attn_causal_f16wmma(const float* __restrict__ Q,
                               const float* __restrict__ K,
                               const float* __restrict__ V,
                               float* __restrict__ O) {
  // K block row-major [key][d], V block transposed [d][key], P^T [query][key]
  __shared__ __align__(16) _Float16 kLds[BN * KP];
  __shared__ __align__(16) _Float16 vLds[kDH * VP];
  __shared__ __align__(16) _Float16 pLds[4 * 16 * PP];

  const int qBlocks = kS / BM;
  const int qb = blockIdx.x % qBlocks;
  const int bh = blockIdx.x / qBlocks;
  const int qbase = qb * BM;
  const size_t bhOff = (size_t)bh * kS * kDH;
  const float* q = Q + bhOff;
  const float* k = K + bhOff;
  const float* v = V + bhOff;
  float* o = O + bhOff;

  const int tid = threadIdx.x;
  const int wave = tid >> 5;
  const int lane = tid & 31;
  const int col = lane & 15;  // N index: this lane's query within the wave tile
  const int hi = lane >> 4;
  const int wloc = wave * 16 + col;  // query position within the 64-row block

  // ---- resident Q^T B-operand (scale*log2e folded in) ----
  // B operand, k order chosen here (contiguous, A matches): lanes 0-15 hold K=0..15, lanes 16-31 K=16..31.
  v16h bq0, bq1;
  {
    const float* qr = q + (size_t)(qbase + wloc) * kDH;
#pragma unroll
    for (int dc = 0; dc < 2; ++dc) {
      const float* p = qr + dc * 32 + hi * 16;
      const float4 f0 = *(const float4*)(p + 0);
      const float4 f1 = *(const float4*)(p + 4);
      const float4 f2 = *(const float4*)(p + 8);
      const float4 f3 = *(const float4*)(p + 12);
      v16h b;
      b[0] = (_Float16)(f0.x * QSCALE);  b[1] = (_Float16)(f0.y * QSCALE);
      b[2] = (_Float16)(f0.z * QSCALE);  b[3] = (_Float16)(f0.w * QSCALE);
      b[4] = (_Float16)(f1.x * QSCALE);  b[5] = (_Float16)(f1.y * QSCALE);
      b[6] = (_Float16)(f1.z * QSCALE);  b[7] = (_Float16)(f1.w * QSCALE);
      b[8] = (_Float16)(f2.x * QSCALE);  b[9] = (_Float16)(f2.y * QSCALE);
      b[10] = (_Float16)(f2.z * QSCALE); b[11] = (_Float16)(f2.w * QSCALE);
      b[12] = (_Float16)(f3.x * QSCALE); b[13] = (_Float16)(f3.y * QSCALE);
      b[14] = (_Float16)(f3.z * QSCALE); b[15] = (_Float16)(f3.w * QSCALE);
      if (dc == 0) bq0 = b; else bq1 = b;
    }
  }

  v8f acc[4] = {v8f{}, v8f{}, v8f{}, v8f{}};  // O^T: dim tiles x 16 queries
  float mrun = -3.0e38f, lrun = 0.0f;         // per-lane (per-query) stats

  // staging assignments
  const int srow = tid >> 1;  // K: key row 0..63
  const int sseg = tid & 1;   // K: which 32-float half of the row
  const int vkg = tid & 15;   // V: key group (4 keys)
  const int vdg = tid >> 4;   // V: dim group (8 dims)

  // A-operand loader (contiguous k order, same as B): lanes 0-15 hold K {o..o+15},
  // lanes 16-31 hold K {o+16..o+31} of their row.
  auto loadA = [&](const _Float16* base, int stride, int row, int kOff) -> v16h {
    const _Float16* p = base + row * stride + kOff + hi * 16;
    return cat8(*(const v8h*)p, *(const v8h*)(p + 8));
  };

  auto stage = [&](int kb, bool pf) {
    // ---- K rows -> f16 row-major (4x ds_store_b128 per thread) ----
    const size_t gk = (size_t)(kb * BN + srow) * kDH + sseg * 32;
    const float4* kr4 = (const float4*)(k + gk);
#pragma unroll
    for (int c = 0; c < 4; ++c) {
      const float4 a0 = kr4[2 * c], a1 = kr4[2 * c + 1];
      v8h h;
      h[0] = (_Float16)a0.x; h[1] = (_Float16)a0.y;
      h[2] = (_Float16)a0.z; h[3] = (_Float16)a0.w;
      h[4] = (_Float16)a1.x; h[5] = (_Float16)a1.y;
      h[6] = (_Float16)a1.z; h[7] = (_Float16)a1.w;
      *(v8h*)&kLds[srow * KP + sseg * 32 + c * 8] = h;
    }
    // ---- V: 4-key x 8-dim register transpose -> [d][key] ----
    // one base pointer; the 4 rows are constant 256B-apart immediate offsets
    const float* vbase = v + (size_t)(kb * BN + vkg * 4) * kDH + vdg * 8;
    float vf[4][8];
#pragma unroll
    for (int kr = 0; kr < 4; ++kr) {
      const float4 b0 = *(const float4*)(vbase + kr * kDH);
      const float4 b1 = *(const float4*)(vbase + kr * kDH + 4);
      vf[kr][0] = b0.x; vf[kr][1] = b0.y; vf[kr][2] = b0.z; vf[kr][3] = b0.w;
      vf[kr][4] = b1.x; vf[kr][5] = b1.y; vf[kr][6] = b1.z; vf[kr][7] = b1.w;
    }
#pragma unroll
    for (int dj = 0; dj < 8; ++dj) {
      v4h hv;
      hv[0] = (_Float16)vf[0][dj]; hv[1] = (_Float16)vf[1][dj];
      hv[2] = (_Float16)vf[2][dj]; hv[3] = (_Float16)vf[3][dj];
      *(v4h*)&vLds[(vdg * 8 + dj) * VP + vkg * 4] = hv;
    }
    if (pf) {  // prefetch next key block into GL2 ahead of its staging pass
      __builtin_prefetch(k + gk + (size_t)BN * kDH, 0, 3);
      __builtin_prefetch(vbase + (size_t)BN * kDH, 0, 3);
    }
  };

  auto compute = [&](bool masked) {
    // ---- S^T = K * Q^T : 4 key M-tiles, K-dim 64 = 2x32 ----
    // C layout: VGPR r <-> key 16t + r + 8*hi, lane col <-> query.
    v8f s[4] = {v8f{}, v8f{}, v8f{}, v8f{}};
#pragma unroll
    for (int t = 0; t < 4; ++t) {
      s[t] = wmma_f16(loadA(kLds, KP, t * 16 + col, 0), bq0, s[t]);
      s[t] = wmma_f16(loadA(kLds, KP, t * 16 + col, 32), bq1, s[t]);
    }
    if (masked) {  // diagonal block only: key(local) > query(local)
#pragma unroll
      for (int t = 0; t < 4; ++t)
#pragma unroll
        for (int r = 0; r < 8; ++r)
          s[t][r] += ((16 * t + r + 8 * hi) > wloc) ? MASK_FILL_LOG2 : 0.0f;
    }
    // ---- online softmax (log2 domain): lane-local + one xor-16 combine ----
    float vmax = -3.0e38f;
#pragma unroll
    for (int t = 0; t < 4; ++t)
#pragma unroll
      for (int r = 0; r < 8; ++r) vmax = fmaxf(vmax, s[t][r]);
    vmax = fmaxf(vmax, __shfl_xor(vmax, 16));
    const float mnew = fmaxf(mrun, vmax);
    const float alpha = __builtin_exp2f(mrun - mnew);
    float psum = 0.0f;
    v8h ph[4];
#pragma unroll
    for (int t = 0; t < 4; ++t)
#pragma unroll
      for (int r = 0; r < 8; ++r) {
        const float p = __builtin_exp2f(s[t][r] - mnew);
        psum += p;
        ph[t][r] = (_Float16)p;
      }
    psum += __shfl_xor(psum, 16);
    lrun = lrun * alpha + psum;
    mrun = mnew;
#pragma unroll
    for (int t = 0; t < 4; ++t)
#pragma unroll
      for (int r = 0; r < 8; ++r) acc[t][r] *= alpha;

    // ---- spill P^T (row = query, contiguous keys per lane: b128 stores) ----
    _Float16* pw = &pLds[(wave * 16 + col) * PP];
#pragma unroll
    for (int t = 0; t < 4; ++t) *(v8h*)&pw[16 * t + 8 * hi] = ph[t];
    asm volatile("" ::: "memory");  // same-wave LDS ops are in-order in HW

    // ---- reload P^T as B-operand, O^T += V^T * P^T ----
    const _Float16* pr = &pLds[(wave * 16 + col) * PP + hi * 16];
    const v16h bp0 = cat8(*(const v8h*)pr, *(const v8h*)(pr + 8));
    const v16h bp1 = cat8(*(const v8h*)(pr + 32), *(const v8h*)(pr + 40));
#pragma unroll
    for (int t = 0; t < 4; ++t) {
      acc[t] = wmma_f16(loadA(vLds, VP, t * 16 + col, 0), bp0, acc[t]);
      acc[t] = wmma_f16(loadA(vLds, VP, t * 16 + col, 32), bp1, acc[t]);
    }
  };

  // ---- main loop: maskless full blocks, then one masked diagonal block ----
  for (int kb = 0; kb < qb; ++kb) {
    __syncthreads();
    stage(kb, true);
    __syncthreads();
    compute(false);
  }
  __syncthreads();
  stage(qb, false);
  __syncthreads();
  compute(true);

  // ---- epilogue: O = acc / l ; per lane: query row, contiguous dims ----
  const float inv = 1.0f / lrun;
  float* orow = o + (size_t)(qbase + wloc) * kDH;
#pragma unroll
  for (int t = 0; t < 4; ++t) {
    float4 u0, u1;
    u0.x = acc[t][0] * inv; u0.y = acc[t][1] * inv;
    u0.z = acc[t][2] * inv; u0.w = acc[t][3] * inv;
    u1.x = acc[t][4] * inv; u1.y = acc[t][5] * inv;
    u1.z = acc[t][6] * inv; u1.w = acc[t][7] * inv;
    *(float4*)(orow + 16 * t + 8 * hi) = u0;
    *(float4*)(orow + 16 * t + 8 * hi + 4) = u1;
  }
}

extern "C" void kernel_launch(void* const* d_in, const int* in_sizes, int n_in,
                              void* d_out, int out_size, void* d_ws, size_t ws_size,
                              hipStream_t stream) {
  (void)in_sizes; (void)n_in; (void)out_size; (void)d_ws; (void)ws_size;
  const float* q = (const float*)d_in[0];
  const float* k = (const float*)d_in[1];
  const float* v = (const float*)d_in[2];
  // d_in[3] is the boolean causal mask; causality is applied analytically
  // (k > q -> bias), identical to the reference additive -10000 bias
  // (scaled by log2(e) since softmax runs in the exp2 domain).
  float* out = (float*)d_out;
  const dim3 grid((kS / BM) * kNB * kH);
  const dim3 block(128);
  hipLaunchKernelGGL(flash_attn_causal_f16wmma, grid, block, 0, stream,
                     q, k, v, out);
}
